// CombinedHiddenGCVAEDecoder_38886633898289
// MI455X (gfx1250) — compile-verified, matrix operand order checked by reading, not run on hardware
//
#include <hip/hip_runtime.h>

typedef __attribute__((ext_vector_type(16))) _Float16 v16h;
typedef __attribute__((ext_vector_type(8)))  float    v8f;

// ---------------------------------------------------------------------------
// Degree / normalization
// ---------------------------------------------------------------------------
__global__ void k_init_deg(float* __restrict__ deg, int n) {
  int i = blockIdx.x * blockDim.x + threadIdx.x;
  if (i < n) deg[i] = 1.0f;  // self loop
}

__global__ void k_count_deg(const int* __restrict__ col, int E, float* __restrict__ deg) {
  int e = blockIdx.x * blockDim.x + threadIdx.x;
  if (e < E) atomicAdd(&deg[col[e]], 1.0f);
}

__global__ void k_rsqrt_inplace(float* __restrict__ d, int n) {
  int i = blockIdx.x * blockDim.x + threadIdx.x;
  if (i < n) d[i] = rsqrtf(d[i]);  // deg >= 1 always (self loop)
}

// ---------------------------------------------------------------------------
// WMMA GEMM:  C[n_rows x n_cols] = A[n_rows x 64] * B[64 x n_cols]
// One wave owns a 16-row stripe: A is loaded and cvt'd to f16 ONCE into two
// v16h fragments (K=0..31 and K=32..63), then the wave loops over all N-tiles
// (B load -> 2x v_wmma_f32_16x16x32_f16 -> store).  4x less A traffic and
// f32->f16 conversion than one-tile-per-wave for n_cols=64.
// A lane layout (16-bit A, 16x32):  m = lane&15 ; K-base = (lane<16?0:8)
//   halves 0..7  -> K = kb+0..7 ;  halves 8..15 -> K = kb+16..23
// B stored per-lane columns (symmetric layout), D: vgpr v holds row
//   16*tm + v + 8*(lane>=16), column = tn*16 + (lane&15).
// n_rows must be a multiple of 16 (100000 = 6250*16).
// ---------------------------------------------------------------------------
__global__ void k_wmma_gemm_k64(const float* __restrict__ A,
                                const float* __restrict__ B,
                                float* __restrict__ C,
                                int n_rows, int n_cols) {
  const int lane = threadIdx.x & 31;
  const int wave = (int)((blockIdx.x * blockDim.x + threadIdx.x) >> 5);  // uniform
  const int tiles_m = n_rows >> 4;
  if (wave >= tiles_m) return;                 // whole-wave exit, EXEC stays all-1

  const int lid  = lane & 15;
  const int half = lane >> 4;                  // 0 or 1
  const int rowA = wave * 16 + lid;            // this lane's A row
  const long abase = (long)rowA * 64;

  // --- load + convert the 16x64 A stripe once: a[0]=K 0..31, a[1]=K 32..63 ---
  v16h a[2];
#pragma unroll
  for (int ks = 0; ks < 2; ++ks) {
    const int kb = ks * 32 + half * 8;
#pragma unroll
    for (int t = 0; t < 8; ++t) {
      a[ks][t]     = (_Float16)A[abase + kb + t];
      a[ks][t + 8] = (_Float16)A[abase + kb + 16 + t];
    }
  }

  const int tiles_n = n_cols >> 4;
  for (int tn = 0; tn < tiles_n; ++tn) {
    const int colB = tn * 16 + lid;            // this lane's B column
    v8f acc = {};
#pragma unroll
    for (int ks = 0; ks < 2; ++ks) {
      const int kb = ks * 32 + half * 8;
      v16h b;
#pragma unroll
      for (int t = 0; t < 8; ++t) {
        b[t]     = (_Float16)B[(long)(kb + t) * n_cols + colB];
        b[t + 8] = (_Float16)B[(long)(kb + 16 + t) * n_cols + colB];
      }
      acc = __builtin_amdgcn_wmma_f32_16x16x32_f16(
          /*neg_a=*/false, a[ks], /*neg_b=*/false, b,
          /*c_mod=*/(short)0, acc, /*reuse_a=*/false, /*reuse_b=*/false);
    }
#pragma unroll
    for (int v = 0; v < 8; ++v) {
      const int m = wave * 16 + v + 8 * half;
      C[(long)m * n_cols + colB] = acc[v];
    }
  }
}

// ---------------------------------------------------------------------------
// out[i,:] = dinv[i]^2 * xw[i,:]   (self-loop term; also initializes out)
// ---------------------------------------------------------------------------
__global__ void k_self_init(const float* __restrict__ xw,
                            const float* __restrict__ dinv,
                            float* __restrict__ out, long total, int shift) {
  long idx = (long)blockIdx.x * blockDim.x + threadIdx.x;
  if (idx >= total) return;
  const float di = dinv[idx >> shift];
  out[idx] = di * di * xw[idx];
}

// ---------------------------------------------------------------------------
// Edge scatter: out[col,:] += dinv[row]*dinv[col] * xw[row,:]
// One thread handles one edge x 4 contiguous dims (float4 gather from L2,
// 4x global_atomic_add_f32 scatter). gshift = log2(dim/4).
// ---------------------------------------------------------------------------
__global__ void k_scatter(const int* __restrict__ row, const int* __restrict__ col,
                          const float* __restrict__ dinv,
                          const float* __restrict__ xw,
                          float* __restrict__ out,
                          long nwork, int dim, int gshift) {
  long idx = (long)blockIdx.x * blockDim.x + threadIdx.x;
  if (idx >= nwork) return;
  const long e = idx >> gshift;
  const int  g = (int)(idx & ((1 << gshift) - 1));
  const int  r = row[e];
  const int  c = col[e];
  const float coef = dinv[r] * dinv[c];
  const float4 v = ((const float4*)(xw + (long)r * dim))[g];
  float* o = out + (long)c * dim + g * 4;
  atomicAdd(o + 0, coef * v.x);
  atomicAdd(o + 1, coef * v.y);
  atomicAdd(o + 2, coef * v.z);
  atomicAdd(o + 3, coef * v.w);
}

// ---------------------------------------------------------------------------
// Bias (+ optional ReLU)
// ---------------------------------------------------------------------------
__global__ void k_bias_act(float* __restrict__ x, const float* __restrict__ b,
                           long total, int dmask, int relu) {
  long idx = (long)blockIdx.x * blockDim.x + threadIdx.x;
  if (idx >= total) return;
  float v = x[idx] + b[idx & dmask];
  x[idx] = relu ? fmaxf(v, 0.0f) : v;
}

// ---------------------------------------------------------------------------
// Launch
// Inputs: 0:x[N,64] f32  1:edge_index[2,E] i32  2:W1[64,64]  3:b1[64]
//         4:W2[64,32]    5:b2[32]        Output: [N,32] f32
// ---------------------------------------------------------------------------
extern "C" void kernel_launch(void* const* d_in, const int* in_sizes, int n_in,
                              void* d_out, int out_size, void* d_ws, size_t ws_size,
                              hipStream_t stream) {
  const float* x   = (const float*)d_in[0];
  const int*   ei  = (const int*)d_in[1];
  const float* W1  = (const float*)d_in[2];
  const float* b1  = (const float*)d_in[3];
  const float* W2  = (const float*)d_in[4];
  const float* b2  = (const float*)d_in[5];
  float* out = (float*)d_out;

  const int N = in_sizes[0] / 64;      // 100000
  const int E = in_sizes[1] / 2;       // 800000
  const int HID = 64, OUT = 32;
  const int* row = ei;                 // edge_index[0] = source
  const int* col = ei + E;             // edge_index[1] = target

  // Workspace carve-up (floats), 256B-aligned blocks.
  char* ws = (char*)d_ws;
  float* dinv = (float*)ws;                       ws += ((size_t)N * 4 + 255) & ~255ull;
  float* xw1  = (float*)ws;                       ws += ((size_t)N * HID * 4 + 255) & ~255ull;
  float* h    = (float*)ws;                       ws += ((size_t)N * HID * 4 + 255) & ~255ull;
  float* hw2  = (float*)ws;

  const int TB = 256;
  auto blocks = [](long n, int tb) { return (int)((n + tb - 1) / tb); };

  // --- normalization ---
  k_init_deg<<<blocks(N, TB), TB, 0, stream>>>(dinv, N);
  k_count_deg<<<blocks(E, TB), TB, 0, stream>>>(col, E, dinv);
  k_rsqrt_inplace<<<blocks(N, TB), TB, 0, stream>>>(dinv, N);

  // --- layer 1: xw1 = x @ W1 ; h = D^-1/2 A D^-1/2 xw1 ; relu(h + b1) ---
  {
    long waves = (long)(N / 16);                  // one wave per 16-row stripe
    k_wmma_gemm_k64<<<blocks(waves * 32, TB), TB, 0, stream>>>(x, W1, xw1, N, HID);
  }
  k_self_init<<<blocks((long)N * HID, TB), TB, 0, stream>>>(xw1, dinv, h, (long)N * HID, 6);
  k_scatter<<<blocks((long)E * (HID / 4), TB), TB, 0, stream>>>(
      row, col, dinv, xw1, h, (long)E * (HID / 4), HID, 4);
  k_bias_act<<<blocks((long)N * HID, TB), TB, 0, stream>>>(h, b1, (long)N * HID, HID - 1, 1);

  // --- layer 2: hw2 = h @ W2 ; out = D^-1/2 A D^-1/2 hw2 + b2 ---
  {
    long waves = (long)(N / 16);
    k_wmma_gemm_k64<<<blocks(waves * 32, TB), TB, 0, stream>>>(h, W2, hw2, N, OUT);
  }
  k_self_init<<<blocks((long)N * OUT, TB), TB, 0, stream>>>(hw2, dinv, out, (long)N * OUT, 5);
  k_scatter<<<blocks((long)E * (OUT / 4), TB), TB, 0, stream>>>(
      row, col, dinv, hw2, out, (long)E * (OUT / 4), OUT, 3);
  k_bias_act<<<blocks((long)N * OUT, TB), TB, 0, stream>>>(out, b2, (long)N * OUT, OUT - 1, 0);
}
